// RNNCFModel_16080357556340
// MI455X (gfx1250) — matrix operand order repaired and checked by reading, not compiled, not run on hardware
//
#include <hip/hip_runtime.h>

// CDNA5 / gfx1250 wave32 WMMA implementation of the LSTM car-following scan.
// One wave = 16 vehicles, full 256-step time loop on-chip, no cross-wave sync
// inside the loop (all loop LDS traffic is wave-private; LDS is in-order per wave).
// Gate GEMM: [x|h|pad](16x32 f16) @ [W;U;0](32x80 f16) via 5x v_wmma_f32_16x16x32_f16.

typedef __attribute__((ext_vector_type(16))) _Float16 v16h;
typedef __attribute__((ext_vector_type(8)))  _Float16 v8h;
typedef __attribute__((ext_vector_type(8)))  float    v8f;

#define NT   256
#define HDIM 20
#define WAVES_PER_BLOCK 8
#define VEH_PER_WAVE    16
#define ZS   84   // padded f32 row stride for z scatter (avoids 2-way bank conflicts)

// Wave-local LDS ordering fence: LDS ops from one wave execute in order; this
// just stops the compiler reordering across it and drains DScnt.
#define WAVE_LDS_FENCE() asm volatile("s_wait_dscnt 0" ::: "memory")

// gfx1250 hardware tanh (TRANS op) when the toolchain exposes it.
#if __has_builtin(__builtin_amdgcn_tanhf)
__device__ __forceinline__ float tanh_(float x) { return __builtin_amdgcn_tanhf(x); }
__device__ __forceinline__ float sigm_(float x) {
    return __builtin_fmaf(0.5f, __builtin_amdgcn_tanhf(0.5f * x), 0.5f);
}
#elif __has_builtin(__builtin_amdgcn_tanh_f32)
__device__ __forceinline__ float tanh_(float x) { return __builtin_amdgcn_tanh_f32(x); }
__device__ __forceinline__ float sigm_(float x) {
    return __builtin_fmaf(0.5f, __builtin_amdgcn_tanh_f32(0.5f * x), 0.5f);
}
#else
__device__ __forceinline__ float sigm_(float x) {
    return __builtin_amdgcn_rcpf(1.0f + __expf(-x));
}
__device__ __forceinline__ float tanh_(float x) {
    return 2.0f * __builtin_amdgcn_rcpf(1.0f + __expf(-2.0f * x)) - 1.0f;
}
#endif

__global__ __launch_bounds__(256, 1) void rnncf_kernel(
    const float* __restrict__ lead,        // [NVEH][NT][2]
    const float* __restrict__ init_state,  // [NVEH][2]  (pos, speed)
    const float* __restrict__ h0,          // [NVEH][H]
    const float* __restrict__ c0,          // [NVEH][H]
    const float* __restrict__ W,           // [3][80]
    const float* __restrict__ U,           // [H][80]
    const float* __restrict__ b,           // [80]
    const float* __restrict__ Wd,          // [H]
    const float* __restrict__ bd,          // [1]
    float* __restrict__ out,               // outputs(NT*NVEH) | speed(NVEH) | h | c
    int nveh)
{
    __shared__ _Float16 sWcat[32 * 80];                    // packed [K=32][N=80] f16 weights
    __shared__ _Float16 sXH[WAVES_PER_BLOCK][16 * 32];     // per-wave A staging [M=16][K=32]
    __shared__ float    sZ[WAVES_PER_BLOCK][16 * ZS];      // per-wave z scatter [M=16][N<=80]

    const int tid  = threadIdx.x;
    const int lane = tid & 31;
    const int w    = tid >> 5;

    // ---- stage concatenated f16 weights: rows 0-2 = W, 3-22 = U, 23-31 = zero pad ----
    for (int idx = tid; idx < 32 * 80; idx += 256) {
        int k = idx / 80, n = idx - k * 80;
        float v = 0.0f;
        if (k < 3)             v = W[k * 80 + n];
        else if (k < 3 + HDIM) v = U[(k - 3) * 80 + n];
        sWcat[idx] = (_Float16)v;
    }
    __syncthreads();   // only cross-wave sync in the kernel

    const int waveBase = (blockIdx.x * WAVES_PER_BLOCK + w) * VEH_PER_WAVE;
    if (waveBase >= nveh) return;   // terminated waves release barrier membership

    const int nloc   = lane & 15;   // A row / D column-in-tile / vehicle for lanes 0-15
    const int hiHalf = lane >> 4;   // 0: lanes 0-15, 1: lanes 16-31

    // ---- loop-invariant B fragments (5 N-tiles) + bias, contiguous K order, same as A:
    //      lane<16: col = n0+lane, K=0..15 ; lane>=16: col = n0+lane-16, K=16..31
    v16h bfrag[5];
    float bias[5];
    #pragma unroll
    for (int t5 = 0; t5 < 5; ++t5) {
        const int col   = t5 * 16 + nloc;
        const int kbase = hiHalf * 16;
        #pragma unroll
        for (int e = 0; e < 16; ++e)
            bfrag[t5][e] = sWcat[(kbase + e) * 80 + col];
        bias[t5] = b[col];
    }

    // ---- gate-lane assignment: 320 (veh, j) pairs -> 10 per lane
    const int vloc  = lane >> 1;            // vehicle 0..15
    const int jbase = (lane & 1) * 10;      // hidden offset 0 or 10
    const int vgGate = waveBase + vloc;
    float wd[10], hreg[10], creg[10];
    #pragma unroll
    for (int q = 0; q < 10; ++q) {
        wd[q]   = Wd[jbase + q];
        hreg[q] = h0[(size_t)vgGate * HDIM + jbase + q];
        creg[q] = c0[(size_t)vgGate * HDIM + jbase + q];
    }

    // ---- init A staging. Zero only K=0..2 and the pad K=23..31 (disjoint from the
    //      h region K=3..22 that the gate lanes seed, so no intra-wave ordering needed).
    _Float16* xh = &sXH[w][0];
    if (lane < 16) {
        #pragma unroll
        for (int k = 0; k < 3; ++k)   xh[nloc * 32 + k] = (_Float16)0.0f;
        #pragma unroll
        for (int k = 23; k < 32; ++k) xh[nloc * 32 + k] = (_Float16)0.0f;
    }
    #pragma unroll
    for (int q = 0; q < 10; ++q)
        xh[vloc * 32 + 3 + jbase + q] = (_Float16)hreg[q];

    // ---- per-vehicle scalar state (lanes 0-15)
    const int vgLane = waveBase + nloc;
    const float bdv = bd[0];
    float pos = 0.0f, speed = 0.0f;
    if (lane < 16) {
        pos   = init_state[(size_t)vgLane * 2 + 0];
        speed = init_state[(size_t)vgLane * 2 + 1];
    }

    // A fragment source: lane<16 -> K [0..15]; lane>=16 -> K [16..31]
    const v8h* aptr0 = (const v8h*)&xh[nloc * 32 + hiHalf * 16];
    const v8h* aptr1 = (const v8h*)&xh[nloc * 32 + hiHalf * 16 + 8];
    float* zrow = &sZ[w][0];

    for (int t = 0; t < NT; ++t) {
        // ---- stage x features for this step
        if (lane < 16) {
            const float2 lv = *(const float2*)&lead[(size_t)(vgLane * NT + t) * 2];
            xh[nloc * 32 + 0] = (_Float16)((lv.x - pos) * (1.0f / 100.0f)); // curhd
            xh[nloc * 32 + 1] = (_Float16)(speed * (1.0f / 40.0f));         // norm_v
            xh[nloc * 32 + 2] = (_Float16)(lv.y  * (1.0f / 40.0f));         // lead_v
            if (t + 8 < NT)
                __builtin_prefetch(&lead[(size_t)(vgLane * NT + t + 8) * 2], 0, 0);
        }
        WAVE_LDS_FENCE();  // x stores + prev-step h stores ordered before A loads

        // ---- A fragment (two b128 LDS chunks per lane)
        v8h ch0 = *aptr0, ch1 = *aptr1;
        v16h afrag;
        #pragma unroll
        for (int e = 0; e < 8; ++e) { afrag[e] = ch0[e]; afrag[e + 8] = ch1[e]; }

        // ---- 5x WMMA over the 80 gate columns, bias-seeded f32 accumulate, scatter z
        #pragma unroll
        for (int t5 = 0; t5 < 5; ++t5) {
            v8f acc;
            #pragma unroll
            for (int r = 0; r < 8; ++r) acc[r] = bias[t5];
            acc = __builtin_amdgcn_wmma_f32_16x16x32_f16(
                      false, afrag, false, bfrag[t5], (short)0, acc, false, false);
            #pragma unroll
            for (int r = 0; r < 8; ++r)   // D layout: lane<16 -> M=r ; lane>=16 -> M=r+8
                zrow[(hiHalf * 8 + r) * ZS + t5 * 16 + nloc] = acc[r];
        }
        WAVE_LDS_FENCE();  // z scatter ordered before cross-lane gate reads

        // ---- LSTM gate math: 10 (veh, j) pairs per lane + dense-head partial dot
        float partial = 0.0f;
        #pragma unroll
        for (int q = 0; q < 10; ++q) {
            const float* zv = &zrow[vloc * ZS + jbase + q];
            float zi = zv[0], zf = zv[20], zg = zv[40], zo = zv[60];
            float cc = sigm_(zf) * creg[q] + sigm_(zi) * tanh_(zg);
            float hh = sigm_(zo) * tanh_(cc);
            creg[q] = cc; hreg[q] = hh;
            xh[vloc * 32 + 3 + jbase + q] = (_Float16)hh;  // feed next step's A
            partial += hh * wd[q];
        }

        // ---- reduce head partials (vehicle v lives in lanes 2v, 2v+1) via ds_bpermute
        float p0 = __shfl(partial, 2 * nloc,     32);
        float p1 = __shfl(partial, 2 * nloc + 1, 32);
        if (lane < 16) {
            float dout = p0 + p1 + bdv;
            float accl = 7.0f * dout - 4.0f;        // (MAXA-MINA)*out + MINA
            speed += 0.1f * accl;
            pos   += 0.1f * accl;
            out[(size_t)t * nveh + vgLane] = pos;   // ys is (NT, NVEH) row-major
        }
    }

    // ---- final state outputs: speed | h | c
    if (lane < 16)
        out[(size_t)NT * nveh + vgLane] = speed;
    const size_t hoff = (size_t)NT * nveh + nveh;
    const size_t coff = hoff + (size_t)nveh * HDIM;
    #pragma unroll
    for (int q = 0; q < 10; ++q) {
        out[hoff + (size_t)vgGate * HDIM + jbase + q] = hreg[q];
        out[coff + (size_t)vgGate * HDIM + jbase + q] = creg[q];
    }
}

extern "C" void kernel_launch(void* const* d_in, const int* in_sizes, int n_in,
                              void* d_out, int out_size, void* d_ws, size_t ws_size,
                              hipStream_t stream) {
    const float* lead       = (const float*)d_in[0];
    const float* init_state = (const float*)d_in[1];
    const float* h0         = (const float*)d_in[2];
    const float* c0         = (const float*)d_in[3];
    const float* W          = (const float*)d_in[4];
    const float* U          = (const float*)d_in[5];
    const float* b          = (const float*)d_in[6];
    const float* Wd         = (const float*)d_in[7];
    const float* bd         = (const float*)d_in[8];
    float* out = (float*)d_out;

    const int nveh = in_sizes[1] / 2;                       // init_state = NVEH*2
    const int vehPerBlock = WAVES_PER_BLOCK * VEH_PER_WAVE; // 128
    const int blocks = (nveh + vehPerBlock - 1) / vehPerBlock;

    rnncf_kernel<<<blocks, 256, 0, stream>>>(lead, init_state, h0, c0,
                                             W, U, b, Wd, bd, out, nveh);
}
